// DeepBKT_51049981281447
// MI455X (gfx1250) — matrix operand order repaired and checked by reading, not compiled, not run on hardware
//
#include <hip/hip_runtime.h>

// ---------------- problem constants ----------------
constexpr int kB   = 16;
constexpr int kS   = 512;
constexpr int kD   = 512;
constexpr int kH   = 8;
constexpr int kDFF = 2048;
constexpr int kL   = 4;
constexpr int kDK  = 64;          // kD / kH
constexpr int kNT  = kB * kS;     // 8192 token rows
#define NEGV (-1e32f)

// ---------------- WMMA plumbing ----------------
typedef __attribute__((ext_vector_type(16))) __bf16 v16bf;
typedef __attribute__((ext_vector_type(8)))  float  v8f;
typedef unsigned short ushort_t;

struct U8 { unsigned u[8]; };

static __device__ __forceinline__ v16bf as_bf(const U8& x) {
    return __builtin_bit_cast(v16bf, x);
}

static __device__ __forceinline__ v8f wmma_bf16(const U8& a, const U8& b, v8f c) {
    // D = A(16x32 bf16) * B(32x16 bf16) + C(16x16 f32)
    return __builtin_amdgcn_wmma_f32_16x16x32_bf16(
        false, as_bf(a), false, as_bf(b), (short)0, c, false, false);
}

// f32 -> bf16 (round-to-nearest-even), raw bits
static __device__ __forceinline__ ushort_t f2bf(float f) {
    unsigned u = __float_as_uint(f);
    u = u + 0x7FFFu + ((u >> 16) & 1u);
    return (ushort_t)(u >> 16);
}

// A fragment (16x32 bf16): for lane-half hi, dwords 0..3 = halves [hi*8, hi*8+7],
// dwords 4..7 = halves [16+hi*8, 16+hi*8+7] of this lane's row (K-contiguous).
static __device__ __forceinline__ U8 fragA(const ushort_t* row, int hi) {
    uint4 a = *(const uint4*)(row + hi * 8);
    uint4 b = *(const uint4*)(row + 16 + hi * 8);
    U8 f;
    f.u[0] = a.x; f.u[1] = a.y; f.u[2] = a.z; f.u[3] = a.w;
    f.u[4] = b.x; f.u[5] = b.y; f.u[6] = b.z; f.u[7] = b.w;
    return f;
}
// B fragment (32x16 bf16): for lane-half hi, dwords 0..3 = halves [hi*8, hi*8+7],
// dwords 4..7 = halves [16+hi*8, 16+hi*8+7] along K for this lane's column (requires K-major storage).
static __device__ __forceinline__ U8 fragB(const ushort_t* col, int hi) {
    uint4 a = *(const uint4*)(col + hi * 8);
    uint4 b = *(const uint4*)(col + 16 + hi * 8);
    U8 f;
    f.u[0] = a.x; f.u[1] = a.y; f.u[2] = a.z; f.u[3] = a.w;
    f.u[4] = b.x; f.u[5] = b.y; f.u[6] = b.z; f.u[7] = b.w;
    return f;
}

// ---------------- elementwise kernels ----------------
// Transposing convert: src f32 [L][K][N] -> dst bf16 [L][N][K]
__global__ __launch_bounds__(256) void cvtT_bf16_kernel(const float* __restrict__ src,
                                                        ushort_t* __restrict__ dst,
                                                        int K, int N, int total) {
    int i = blockIdx.x * 256 + threadIdx.x;
    if (i >= total) return;
    int kn = K * N;
    int l = i / kn, r = i - l * kn;
    int k = r / N, n = r - k * N;
    dst[(size_t)l * kn + (size_t)n * K + k] = f2bf(src[i]);
}

__global__ __launch_bounds__(256) void addpe_kernel(const float* __restrict__ q,
                                                    const float* __restrict__ qa,
                                                    const float* __restrict__ pe,
                                                    float* __restrict__ xf,
                                                    ushort_t* __restrict__ xb,
                                                    ushort_t* __restrict__ yb) {
    int i = blockIdx.x * 256 + threadIdx.x;
    if (i >= kNT * kD) return;
    int d = i % kD;
    int s = (i / kD) % kS;
    float p  = pe[s * kD + d];
    float xv = q[i] + p;
    float yv = qa[i] + p;
    xf[i] = xv;
    xb[i] = f2bf(xv);
    yb[i] = f2bf(yv);
}

// x = LN(x + delta) * g + b ; refresh f32 master and bf16 shadow
__global__ __launch_bounds__(256) void add_ln_kernel(float* __restrict__ xf,
                                                     ushort_t* __restrict__ xb,
                                                     const float* __restrict__ delta,
                                                     const float* __restrict__ g,
                                                     const float* __restrict__ beta) {
    __shared__ float red[256];
    int row = blockIdx.x, tid = threadIdx.x;
    size_t base = (size_t)row * kD;
    int i0 = tid * 2;
    float t0 = xf[base + i0]     + delta[base + i0];
    float t1 = xf[base + i0 + 1] + delta[base + i0 + 1];
    red[tid] = t0 + t1;
    __syncthreads();
    for (int s = 128; s > 0; s >>= 1) { if (tid < s) red[tid] += red[tid + s]; __syncthreads(); }
    float mu = red[0] * (1.0f / kD);
    __syncthreads();
    float d0 = t0 - mu, d1 = t1 - mu;
    red[tid] = d0 * d0 + d1 * d1;
    __syncthreads();
    for (int s = 128; s > 0; s >>= 1) { if (tid < s) red[tid] += red[tid + s]; __syncthreads(); }
    float rinv = rsqrtf(red[0] * (1.0f / kD) + 1e-5f);
    float o0 = d0 * rinv * g[i0]     + beta[i0];
    float o1 = d1 * rinv * g[i0 + 1] + beta[i0 + 1];
    xf[base + i0]     = o0;  xf[base + i0 + 1] = o1;
    xb[base + i0]     = f2bf(o0);
    xb[base + i0 + 1] = f2bf(o1);
}

// ---------------- bf16 WMMA GEMM: C[M,N] = A[M,K] * W[K,N] + bias, opt ReLU ----------
// W supplied TRANSPOSED (Wt[N][K]).  256 threads = 8 waves; block tile 128M x 128N;
// per wave 32M x 64N (2x4 WMMA accumulators); K-slab 32; double-buffered LDS.
// outB layout: mode 0 = row-major [M][N]; mode 1 = V-transpose [b][h][dk][s].
__global__ __launch_bounds__(256) void gemm_bf16_kernel(const ushort_t* __restrict__ A,
                                                        const ushort_t* __restrict__ Wt,
                                                        const float* __restrict__ bias,
                                                        float* __restrict__ outF,
                                                        ushort_t* __restrict__ outB,
                                                        int M, int N, int K,
                                                        int relu, int vtmode) {
    __shared__ ushort_t As[2][128][40];   // [buf][row][K]   (K-contiguous, +pad)
    __shared__ ushort_t Bt[2][128][40];   // [buf][col][K]   (K-major, +pad)
    const int tid  = threadIdx.x;
    const int wave = tid >> 5, lane = tid & 31;
    const int wm = wave & 3, wn = wave >> 2;
    const int hi = lane >> 4, ln = lane & 15;
    const int m0 = blockIdx.y * 128, n0 = blockIdx.x * 128;

    // staging coordinates: each thread moves 16 halves of A and 16 of B per slab
    const int srow = tid >> 1, shalf = (tid & 1) * 16;
    const ushort_t* aSrc = A  + (size_t)(m0 + srow) * K + shalf;
    const ushort_t* bSrc = Wt + (size_t)(n0 + srow) * K + shalf;

    v8f acc[2][4];
    #pragma unroll
    for (int mi = 0; mi < 2; ++mi)
        #pragma unroll
        for (int t = 0; t < 4; ++t)
            acc[mi][t] = v8f{0.f, 0.f, 0.f, 0.f, 0.f, 0.f, 0.f, 0.f};

    const int nk = K >> 5;
    // prologue: stage slab 0 into buffer 0
    {
        const uint4* sa = (const uint4*)aSrc;
        const uint4* sb = (const uint4*)bSrc;
        uint4 a0 = sa[0], a1 = sa[1], b0 = sb[0], b1 = sb[1];
        *(uint4*)&As[0][srow][shalf]     = a0;
        *(uint4*)&As[0][srow][shalf + 8] = a1;
        *(uint4*)&Bt[0][srow][shalf]     = b0;
        *(uint4*)&Bt[0][srow][shalf + 8] = b1;
    }
    for (int ks = 0; ks < nk; ++ks) {
        const int buf = ks & 1;
        __syncthreads();   // staging of `buf` complete; previous reads of `buf^1` complete
        if (ks + 1 < nk) { // stage next slab into the other buffer
            const uint4* sa = (const uint4*)(aSrc + (ks + 1) * 32);
            const uint4* sb = (const uint4*)(bSrc + (ks + 1) * 32);
            uint4 a0 = sa[0], a1 = sa[1], b0 = sb[0], b1 = sb[1];
            *(uint4*)&As[buf ^ 1][srow][shalf]     = a0;
            *(uint4*)&As[buf ^ 1][srow][shalf + 8] = a1;
            *(uint4*)&Bt[buf ^ 1][srow][shalf]     = b0;
            *(uint4*)&Bt[buf ^ 1][srow][shalf + 8] = b1;
        }
        U8 a0 = fragA(&As[buf][wm * 32 + ln][0], hi);
        U8 a1 = fragA(&As[buf][wm * 32 + 16 + ln][0], hi);
        #pragma unroll
        for (int t = 0; t < 4; ++t) {
            U8 bf = fragB(&Bt[buf][wn * 64 + t * 16 + ln][0], hi);
            acc[0][t] = wmma_bf16(a0, bf, acc[0][t]);
            acc[1][t] = wmma_bf16(a1, bf, acc[1][t]);
        }
    }

    #pragma unroll
    for (int mi = 0; mi < 2; ++mi) {
        #pragma unroll
        for (int t = 0; t < 4; ++t) {
            int col = n0 + wn * 64 + t * 16 + ln;
            float bv = bias[col];
            #pragma unroll
            for (int r = 0; r < 8; ++r) {
                int row = m0 + wm * 32 + mi * 16 + r + hi * 8;
                float v = acc[mi][t][r] + bv;
                if (relu) v = fmaxf(v, 0.f);
                if (outF) outF[(size_t)row * N + col] = v;
                if (outB) {
                    if (vtmode) { // [b][h][dk][s]
                        int bb = row >> 9, s = row & (kS - 1);
                        int hh = col >> 6, d = col & (kDK - 1);
                        outB[(((size_t)(bb * kH + hh)) * kDK + d) * kS + s] = f2bf(v);
                    } else {
                        outB[(size_t)row * N + col] = f2bf(v);
                    }
                }
            }
        }
    }
}

// ---------------- attention: per (b,h,16-query-row tile) ----------------
// scores = (Q Q^T) * scale * fr[b,i]; strict-lower mask; softmax; row0 zero; out = attn @ V
// qk: bf16 [b*S+s][D]; vt: bf16 [b][h][dk][s]; out: bf16 [b*S+s][D]
__global__ __launch_bounds__(128) void attn_kernel(const ushort_t* __restrict__ qk,
                                                   const ushort_t* __restrict__ vt,
                                                   const float* __restrict__ fr,
                                                   ushort_t* __restrict__ out) {
    __shared__ float    sc[16][516];   // 16 x 512 scores (f32)
    __shared__ ushort_t atn[16][512];  // 16 x 512 attn (bf16)
    __shared__ float    red[16][8];

    const int q0 = blockIdx.x * 16;
    const int h  = blockIdx.y;
    const int b  = blockIdx.z;
    const int tid = threadIdx.x;
    const int wave = tid >> 5, lane = tid & 31;
    const int hi = lane >> 4, ln = lane & 15;
    const float scale = 0.125f; // 1/sqrt(64)
    const size_t hcol = (size_t)h * kDK;

    // ---- phase 1: scores (only tiles at/below the diagonal) ----
    U8 aF[2];
    {
        const ushort_t* qrp = qk + ((size_t)(b * kS + q0 + ln)) * kD + hcol;
        aF[0] = fragA(qrp, hi);
        aF[1] = fragA(qrp + 32, hi);
    }
    const int jtmax = q0 >> 4;             // last key tile with any unmasked entry
    for (int jt = wave; jt <= jtmax; jt += 4) {
        v8f c = v8f{0.f, 0.f, 0.f, 0.f, 0.f, 0.f, 0.f, 0.f};
        int j = jt * 16 + ln;              // key index covered by this lane
        const ushort_t* krp = qk + ((size_t)(b * kS + j)) * kD + hcol;
        c = wmma_bf16(aF[0], fragB(krp, hi), c);
        c = wmma_bf16(aF[1], fragB(krp + 32, hi), c);
        #pragma unroll
        for (int r = 0; r < 8; ++r) {
            int m = r + hi * 8;
            int qrow = q0 + m;
            float s = c[r] * scale * fr[(size_t)b * kS + qrow];
            sc[m][j] = (j < qrow) ? s : NEGV;
        }
    }
    { // fill fully-masked region j >= (jtmax+1)*16
        int jf = (jtmax + 1) * 16;
        int nf = kS - jf;
        for (int idx = tid; idx < 16 * nf; idx += 128) {
            int m = idx / nf, j = jf + idx - m * nf;
            sc[m][j] = NEGV;
        }
    }
    __syncthreads();

    // ---- phase 2: softmax (16 rows x 512, 8 threads per row) ----
    {
        int row = tid >> 3, seg = tid & 7, jb = seg * 64;
        float mx = -3.4e38f;
        for (int t = 0; t < 64; ++t) mx = fmaxf(mx, sc[row][jb + t]);
        red[row][seg] = mx;
        __syncthreads();
        float mxall = red[row][0];
        #pragma unroll
        for (int t = 1; t < 8; ++t) mxall = fmaxf(mxall, red[row][t]);
        float sum = 0.f;
        for (int t = 0; t < 64; ++t) {
            float e = __expf(sc[row][jb + t] - mxall);
            sc[row][jb + t] = e;
            sum += e;
        }
        __syncthreads();
        red[row][seg] = sum;
        __syncthreads();
        float tot = 0.f;
        #pragma unroll
        for (int t = 0; t < 8; ++t) tot += red[row][t];
        float inv = (q0 + row == 0) ? 0.f : (1.0f / tot);   // attn[:, :, 0, :] = 0
        for (int t = 0; t < 64; ++t) atn[row][jb + t] = f2bf(sc[row][jb + t] * inv);
    }
    __syncthreads();

    // ---- phase 3: out = attn @ V, each wave owns one 16-wide n-tile ----
    {
        v8f c = v8f{0.f, 0.f, 0.f, 0.f, 0.f, 0.f, 0.f, 0.f};
        int n = wave * 16 + ln;            // head-dim column of this lane
        const ushort_t* vcol = vt + (((size_t)(b * kH + h)) * kDK + n) * kS;
        const int ktmax = (q0 + 15) >> 5;  // attn is zero beyond key q0+15
        for (int kt = 0; kt <= ktmax; ++kt) {
            U8 a2 = fragA(&atn[ln][kt * 32], hi);
            U8 b2 = fragB(vcol + kt * 32, hi);
            c = wmma_bf16(a2, b2, c);
        }
        #pragma unroll
        for (int r = 0; r < 8; ++r) {
            int qrow = q0 + r + hi * 8;
            out[((size_t)(b * kS + qrow)) * kD + hcol + n] = f2bf(c[r]);
        }
    }
}

// ---------------- host launcher ----------------
extern "C" void kernel_launch(void* const* d_in, const int* in_sizes, int n_in,
                              void* d_out, int out_size, void* d_ws, size_t ws_size,
                              hipStream_t stream) {
    const float* q_embed = (const float*)d_in[0];
    const float* qa_embed= (const float*)d_in[1];
    const float* fr      = (const float*)d_in[2];
    const float* pe      = (const float*)d_in[3];
    const float* Wk      = (const float*)d_in[4];
    const float* bk      = (const float*)d_in[5];
    const float* Wv      = (const float*)d_in[6];
    const float* bv      = (const float*)d_in[7];
    const float* Wo      = (const float*)d_in[8];
    const float* bo      = (const float*)d_in[9];
    const float* ln1_g   = (const float*)d_in[10];
    const float* ln1_b   = (const float*)d_in[11];
    const float* W1      = (const float*)d_in[12];
    const float* b1      = (const float*)d_in[13];
    const float* W2      = (const float*)d_in[14];
    const float* b2      = (const float*)d_in[15];
    const float* ln2_g   = (const float*)d_in[16];
    const float* ln2_b   = (const float*)d_in[17];

    char* ws = (char*)d_ws;
    size_t off = 0;
    auto alloc = [&](size_t bytes) { size_t o = off; off = (off + bytes + 255) & ~(size_t)255; return o; };
    const size_t ND = (size_t)kNT * kD;
    float*    xf  = (float*)   (ws + alloc(ND * 4));
    ushort_t* xb  = (ushort_t*)(ws + alloc(ND * 2));
    ushort_t* yb  = (ushort_t*)(ws + alloc(ND * 2));
    ushort_t* qkb = (ushort_t*)(ws + alloc(ND * 2));
    ushort_t* vtb = (ushort_t*)(ws + alloc(ND * 2));   // V transposed [b][h][dk][s]
    ushort_t* aob = (ushort_t*)(ws + alloc(ND * 2));
    float*    pf  = (float*)   (ws + alloc(ND * 4));
    ushort_t* h1b = (ushort_t*)(ws + alloc((size_t)kNT * kDFF * 2));
    ushort_t* wkb = (ushort_t*)(ws + alloc((size_t)kL * kD * kD * 2));      // transposed
    ushort_t* wvb = (ushort_t*)(ws + alloc((size_t)kL * kD * kD * 2));      // transposed
    ushort_t* wob = (ushort_t*)(ws + alloc((size_t)kL * kD * kD * 2));      // transposed
    ushort_t* w1b = (ushort_t*)(ws + alloc((size_t)kL * kD * kDFF * 2));    // transposed
    ushort_t* w2b = (ushort_t*)(ws + alloc((size_t)kL * kDFF * kD * 2));    // transposed

    auto cvtT = [&](const float* s, ushort_t* d, int K, int N) {
        int total = kL * K * N;
        cvtT_bf16_kernel<<<(total + 255) / 256, 256, 0, stream>>>(s, d, K, N, total);
    };
    cvtT(Wk, wkb, kD, kD);
    cvtT(Wv, wvb, kD, kD);
    cvtT(Wo, wob, kD, kD);
    cvtT(W1, w1b, kD, kDFF);
    cvtT(W2, w2b, kDFF, kD);

    addpe_kernel<<<(kNT * kD + 255) / 256, 256, 0, stream>>>(q_embed, qa_embed, pe, xf, xb, yb);

    auto gemm = [&](const ushort_t* A, const ushort_t* Wt, const float* bias,
                    float* oF, ushort_t* oB, int M, int N, int K, int relu, int vtmode) {
        dim3 grid(N / 128, M / 128);
        gemm_bf16_kernel<<<grid, 256, 0, stream>>>(A, Wt, bias, oF, oB, M, N, K, relu, vtmode);
    };

    for (int l = 0; l < kL; ++l) {
        const size_t wDD = (size_t)l * kD * kD;
        gemm(xb, wkb + wDD, bk + (size_t)l * kD, nullptr, qkb, kNT, kD, kD, 0, 0);
        gemm(yb, wvb + wDD, bv + (size_t)l * kD, nullptr, vtb, kNT, kD, kD, 0, 1);
        attn_kernel<<<dim3(kS / 16, kH, kB), 128, 0, stream>>>(qkb, vtb, fr, aob);
        gemm(aob, wob + wDD, bo + (size_t)l * kD, pf, nullptr, kNT, kD, kD, 0, 0);
        add_ln_kernel<<<kNT, 256, 0, stream>>>(xf, xb, pf, ln1_g + (size_t)l * kD, ln1_b + (size_t)l * kD);
        gemm(xb, w1b + (size_t)l * kD * kDFF, b1 + (size_t)l * kDFF, nullptr, h1b, kNT, kDFF, kD, 1, 0);
        gemm(h1b, w2b + (size_t)l * kDFF * kD, b2 + (size_t)l * kD, pf, nullptr, kNT, kD, kDFF, 0, 0);
        add_ln_kernel<<<kNT, 256, 0, stream>>>(xf, xb, pf, ln2_g + (size_t)l * kD, ln2_b + (size_t)l * kD);
    }

    hipMemcpyAsync(d_out, xf, ND * sizeof(float), hipMemcpyDeviceToDevice, stream);
}
